// GCN_OVER_MLP_9887014715848
// MI455X (gfx1250) — matrix operand order repaired and checked by reading, not compiled, not run on hardware
//
#include <hip/hip_runtime.h>
#include <hip/hip_bf16.h>

// ---------------------------------------------------------------------------
// GCN over MLP for MI455X (gfx1250, wave32, WMMA).
// Heavy GEMMs run on v_wmma_f32_16x16x32_bf16 with fp32 accumulation.
// fp32 operands are converted to bf16 while staging tiles in LDS; bf16
// A operands are copied global->LDS with GLOBAL_LOAD_ASYNC_TO_LDS_B128.
// ---------------------------------------------------------------------------

#define NNODES 4096
#define NBATCH 32
#define FTOT   193
#define DTDIM  128
#define HDIM   256
#define CDIM   64

typedef __attribute__((ext_vector_type(16))) __bf16 v16bf;
typedef __attribute__((ext_vector_type(8)))  float  v8f;

union FragU {
    uint4 q[2];   // two 16-byte LDS loads
    v16bf b;      // WMMA operand view (32 bytes)
};

__device__ __forceinline__ unsigned short f2bf(float f) {
    unsigned int u = __float_as_uint(f);
    u += 0x7FFFu + ((u >> 16) & 1u);          // round-to-nearest-even
    return (unsigned short)(u >> 16);
}
__device__ __forceinline__ float bf2f(unsigned short v) {
    return __uint_as_float(((unsigned int)v) << 16);
}
__device__ __forceinline__ unsigned short toBF(float f)          { return f2bf(f); }
__device__ __forceinline__ unsigned short toBF(unsigned short v) { return v; }

// ---------------------------------------------------------------------------
// bf16 WMMA GEMM:  D(bf16) = [relu]( A(MxK) * B(KxN) [+ bias] )
//   AT/BT: float (converted to bf16 in LDS) or unsigned short (already bf16).
//   AVEC:  A rows are 16-byte aligned/vectorizable (true for adj & bf16 acts).
//   Block tile 128(M) x 64(N), K-step 32. 8 waves = 4(M) x 2(N),
//   each wave computes 32x32 via 4 accumulators of 16x16.
//   Requires M%128==0, N%64==0, K%32==0 (all shapes here satisfy this).
// ---------------------------------------------------------------------------
template <typename AT, typename BT, bool AVEC, bool BIAS, bool RELU>
__global__ __launch_bounds__(256)
void gemm_bf16_wmma(const AT* __restrict__ A,  int lda, long long strideA,
                    const BT* __restrict__ Bm, int ldb, long long strideB,
                    unsigned short* __restrict__ D, int ldd, long long strideD,
                    const float* __restrict__ bias, int K)
{
    __shared__ __align__(16) unsigned short smem[8192];                 // 16 KB
    unsigned short (*lA)[32]  = (unsigned short (*)[32])smem;           // 128x32
    unsigned short (*lBt)[32] = (unsigned short (*)[32])(smem + 4096);  // 64x32 (B transposed)
    unsigned short (*lD)[64]  = (unsigned short (*)[64])smem;           // 128x64 (epilogue, aliases)

    const int t    = threadIdx.x;
    const int lane = t & 31;
    const int wid  = t >> 5;
    const int wy   = wid & 3;          // wave row (4 tiles of 32 rows)
    const int wx   = wid >> 2;         // wave col (2 tiles of 32 cols)

    const int rowBlk = blockIdx.y * 128;
    const int colBlk = blockIdx.x * 64;
    const int z      = blockIdx.z;

    const AT* Ab = A  + (size_t)z * strideA;
    const BT* Bb = Bm + (size_t)z * strideB;
    unsigned short* Db = D + (size_t)z * strideD;

    // staging geometry: A -> thread covers (row aR, cols aC..aC+15)
    //                   B -> thread covers (row bR, cols bC..bC+7)
    const int aR = t >> 1, aC = (t & 1) * 16;
    const int bR = t >> 3, bC = (t & 7) * 8;

    float bias0 = 0.0f, bias1 = 0.0f;
    if (BIAS) {
        const int bcol = colBlk + wx * 32 + (lane & 15);
        bias0 = bias[bcol];
        bias1 = bias[bcol + 16];
    }

    v8f acc[2][2] = {};

    for (int k0 = 0; k0 < K; k0 += 32) {
        // ---- stage A tile 128x32 ----
        const size_t aOff = (size_t)(rowBlk + aR) * lda + (k0 + aC);
        if constexpr (sizeof(AT) == 2 && AVEC) {
            // bf16 source, 16B aligned: async copy straight into LDS (CDNA5)
            const unsigned ldsDst = (unsigned)((const char*)&lA[aR][aC] - (const char*)smem);
            const unsigned short* g = (const unsigned short*)&Ab[aOff];
            asm volatile("global_load_async_to_lds_b128 %0, %1, off"
                         :: "v"(ldsDst), "v"(g) : "memory");
            asm volatile("global_load_async_to_lds_b128 %0, %1, off offset:16"
                         :: "v"(ldsDst), "v"(g) : "memory");
        } else if constexpr (AVEC) {
            // fp32 source, 16B aligned: vector loads + pack to bf16
            const float4* g = (const float4*)&Ab[aOff];
            #pragma unroll
            for (int j = 0; j < 4; ++j) {
                float4 f = g[j];
                uint2 pk;
                pk.x = ((unsigned)toBF(f.y) << 16) | toBF(f.x);
                pk.y = ((unsigned)toBF(f.w) << 16) | toBF(f.z);
                *(uint2*)&lA[aR][aC + j * 4] = pk;
            }
        } else {
            // fp32 source, unaligned rows (x, lda=193): scalar loads
            #pragma unroll
            for (int j = 0; j < 16; ++j)
                lA[aR][aC + j] = toBF(Ab[aOff + j]);
        }

        // ---- stage B tile 32x64, transposed so columns are contiguous ----
        const size_t bOff = (size_t)(k0 + bR) * ldb + (colBlk + bC);
        if constexpr (sizeof(BT) == 2) {
            union { uint4 q; unsigned short h[8]; } u;
            u.q = *(const uint4*)&Bb[bOff];
            #pragma unroll
            for (int j = 0; j < 8; ++j) lBt[bC + j][bR] = u.h[j];
        } else {
            union { float4 f; float s[4]; } u0, u1;
            u0.f = *(const float4*)&Bb[bOff];
            u1.f = *(const float4*)&Bb[bOff + 4];
            #pragma unroll
            for (int j = 0; j < 4; ++j) {
                lBt[bC + j][bR]     = toBF(u0.s[j]);
                lBt[bC + 4 + j][bR] = toBF(u1.s[j]);
            }
        }

        if constexpr (sizeof(AT) == 2 && AVEC)
            asm volatile("s_wait_asynccnt 0" ::: "memory");
        __syncthreads();

        // ---- build fragments per ISA 7.12.2 layouts ----
        const int arow = wy * 32 + (lane & 15);
        const int kbA  = (lane >> 4) * 8;     // lanes 0-15: K 0-7/16-23; 16-31: K 8-15/24-31
        FragU a0, a1;
        a0.q[0] = *(const uint4*)&lA[arow][kbA];
        a0.q[1] = *(const uint4*)&lA[arow][kbA + 16];
        a1.q[0] = *(const uint4*)&lA[arow + 16][kbA];
        a1.q[1] = *(const uint4*)&lA[arow + 16][kbA + 16];

        const int bcol = wx * 32 + (lane & 15);
        const int kbB  = (lane >> 4) * 8;     // lanes 0-15: K 0-7/16-23; 16-31: K 8-15/24-31
        FragU b0, b1;
        b0.q[0] = *(const uint4*)&lBt[bcol][kbB];
        b0.q[1] = *(const uint4*)&lBt[bcol][kbB + 16];
        b1.q[0] = *(const uint4*)&lBt[bcol + 16][kbB];
        b1.q[1] = *(const uint4*)&lBt[bcol + 16][kbB + 16];

        acc[0][0] = __builtin_amdgcn_wmma_f32_16x16x32_bf16(false, a0.b, false, b0.b,
                                                            (short)0, acc[0][0], false, false);
        acc[0][1] = __builtin_amdgcn_wmma_f32_16x16x32_bf16(false, a0.b, false, b1.b,
                                                            (short)0, acc[0][1], false, false);
        acc[1][0] = __builtin_amdgcn_wmma_f32_16x16x32_bf16(false, a1.b, false, b0.b,
                                                            (short)0, acc[1][0], false, false);
        acc[1][1] = __builtin_amdgcn_wmma_f32_16x16x32_bf16(false, a1.b, false, b1.b,
                                                            (short)0, acc[1][1], false, false);
        __syncthreads();
    }

    // ---- epilogue: bias + relu, stage tile in LDS, coalesced b128 stores ----
    #pragma unroll
    for (int my = 0; my < 2; ++my) {
        #pragma unroll
        for (int nx = 0; nx < 2; ++nx) {
            const float badd = BIAS ? (nx ? bias1 : bias0) : 0.0f;
            #pragma unroll
            for (int v = 0; v < 8; ++v) {
                float val = acc[my][nx][v] + badd;
                if (RELU) val = fmaxf(val, 0.0f);
                const int r = wy * 32 + my * 16 + v + ((lane >> 4) * 8);
                const int c = wx * 32 + nx * 16 + (lane & 15);
                lD[r][c] = f2bf(val);
            }
        }
    }
    __syncthreads();
    #pragma unroll
    for (int j = 0; j < 4; ++j) {
        const int chunk = t * 4 + j;           // 1024 chunks of 8 bf16
        const int r  = chunk >> 3;
        const int c8 = (chunk & 7) * 8;
        *(uint4*)&Db[(size_t)(rowBlk + r) * ldd + colBlk + c8] = *(const uint4*)&lD[r][c8];
    }
}

// ---------------------------------------------------------------------------
// BatchNorm over the node axis, in place on bf16 activations [B][N][F].
// ---------------------------------------------------------------------------
__global__ __launch_bounds__(256)
void bn_kernel(unsigned short* __restrict__ act, int F)
{
    const int f = blockIdx.x;
    const int b = blockIdx.y;
    const int t = threadIdx.x;
    __shared__ float s1[256], s2[256];
    __shared__ float mv[2];

    unsigned short* base = act + (size_t)b * NNODES * F + f;
    float vals[16];
    float sum = 0.0f, sq = 0.0f;
    #pragma unroll
    for (int i = 0; i < 16; ++i) {
        int n = t + i * 256;
        float v = bf2f(base[(size_t)n * F]);
        vals[i] = v;
        sum += v; sq += v * v;
    }
    s1[t] = sum; s2[t] = sq;
    __syncthreads();
    for (int s = 128; s > 0; s >>= 1) {
        if (t < s) { s1[t] += s1[t + s]; s2[t] += s2[t + s]; }
        __syncthreads();
    }
    if (t == 0) {
        float m   = s1[0] * (1.0f / NNODES);
        float var = s2[0] * (1.0f / NNODES) - m * m;
        mv[0] = m;
        mv[1] = rsqrtf(var + 1e-5f);
    }
    __syncthreads();
    const float m = mv[0], r = mv[1];
    #pragma unroll
    for (int i = 0; i < 16; ++i) {
        int n = t + i * 256;
        base[(size_t)n * F] = f2bf((vals[i] - m) * r);
    }
}

// ---------------------------------------------------------------------------
// denom = count_nonzero(x[0, :, 192])
// ---------------------------------------------------------------------------
__global__ __launch_bounds__(256)
void count_kernel(const float* __restrict__ x, float* __restrict__ denom)
{
    __shared__ float s[256];
    const int t = threadIdx.x;
    float c = 0.0f;
    for (int n = t; n < NNODES; n += 256)
        c += (x[(size_t)n * FTOT + (FTOT - 1)] != 0.0f) ? 1.0f : 0.0f;
    s[t] = c;
    __syncthreads();
    for (int k = 128; k > 0; k >>= 1) {
        if (t < k) s[t] += s[t + k];
        __syncthreads();
    }
    if (t == 0) *denom = s[0];
}

// ---------------------------------------------------------------------------
// pooled[b][f] = sum_n mask[b,n] * (f<64 ? Y3[b,n,f] : x[b,n,128+f-64]) / denom
// ---------------------------------------------------------------------------
__global__ __launch_bounds__(256)
void pool_kernel(const unsigned short* __restrict__ Y3,
                 const float* __restrict__ x,
                 const float* __restrict__ denom,
                 float* __restrict__ pooled)
{
    const int f = blockIdx.x;   // 0..127
    const int b = blockIdx.y;   // 0..31
    const int t = threadIdx.x;
    __shared__ float s[256];
    float sum = 0.0f;
    for (int n = t; n < NNODES; n += 256) {
        size_t rn = (size_t)b * NNODES + n;
        float m = x[rn * FTOT + (FTOT - 1)];
        float v = (f < CDIM) ? bf2f(Y3[rn * CDIM + f])
                             : x[rn * FTOT + DTDIM + (f - CDIM)];
        sum += v * m;
    }
    s[t] = sum;
    __syncthreads();
    for (int k = 128; k > 0; k >>= 1) {
        if (t < k) s[t] += s[t + k];
        __syncthreads();
    }
    if (t == 0) pooled[(size_t)b * 128 + f] = s[0] / (*denom);
}

// ---------------------------------------------------------------------------
// Tiny MLP head: (32x128) -> relu 256 -> relu 128 -> 1. One block per batch.
// ---------------------------------------------------------------------------
__global__ __launch_bounds__(256)
void head_kernel(const float* __restrict__ pooled,
                 const float* __restrict__ LW1, const float* __restrict__ Lb1,
                 const float* __restrict__ LW2, const float* __restrict__ Lb2,
                 const float* __restrict__ LW3, const float* __restrict__ Lb3,
                 float* __restrict__ out)
{
    const int b = blockIdx.x;
    const int t = threadIdx.x;
    __shared__ float p[128], z1[256], z2[128];

    if (t < 128) p[t] = pooled[(size_t)b * 128 + t];
    __syncthreads();

    float s = Lb1[t];
    for (int k = 0; k < 128; ++k) s += p[k] * LW1[(size_t)k * 256 + t];
    z1[t] = fmaxf(s, 0.0f);
    __syncthreads();

    if (t < 128) {
        float s2 = Lb2[t];
        for (int k = 0; k < 256; ++k) s2 += z1[k] * LW2[(size_t)k * 128 + t];
        z2[t] = fmaxf(s2, 0.0f);
    }
    __syncthreads();

    if (t == 0) {
        float s3 = Lb3[0];
        for (int k = 0; k < 128; ++k) s3 += z2[k] * LW3[k];
        out[b] = s3;
    }
}

// ---------------------------------------------------------------------------
// Host-side orchestration
// ---------------------------------------------------------------------------
extern "C" void kernel_launch(void* const* d_in, const int* in_sizes, int n_in,
                              void* d_out, int out_size, void* d_ws, size_t ws_size,
                              hipStream_t stream)
{
    const float* x   = (const float*)d_in[0];
    const float* adj = (const float*)d_in[1];
    const float* W1  = (const float*)d_in[2];
    const float* b1  = (const float*)d_in[3];
    const float* W2  = (const float*)d_in[4];
    const float* b2  = (const float*)d_in[5];
    const float* W3  = (const float*)d_in[6];
    const float* b3  = (const float*)d_in[7];
    const float* LW1 = (const float*)d_in[8];
    const float* Lb1 = (const float*)d_in[9];
    const float* LW2 = (const float*)d_in[10];
    const float* Lb2 = (const float*)d_in[11];
    const float* LW3 = (const float*)d_in[12];
    const float* Lb3 = (const float*)d_in[13];
    float* out = (float*)d_out;

    // workspace: P, Q ping-pong activation buffers (bf16), pooled, denom
    const size_t actElems = (size_t)NBATCH * NNODES * HDIM;        // 33,554,432
    unsigned short* P = (unsigned short*)d_ws;
    unsigned short* Q = P + actElems;
    float* pooled = (float*)(Q + actElems);
    float* denom  = pooled + NBATCH * 128;

    const int M_all = NBATCH * NNODES;          // 131072
    const dim3 blk(256);
    const long long sAct = (long long)NNODES * HDIM;   // per-batch activation stride
    const long long sC   = (long long)NNODES * CDIM;

    // ---- Layer 1: T1 = xt @ W1 ----
    gemm_bf16_wmma<float, float, false, false, false>
        <<<dim3(HDIM / 64, M_all / 128, 1), blk, 0, stream>>>(
        x, FTOT, 0, W1, HDIM, 0, P, HDIM, 0, nullptr, DTDIM);
    // Y1 = relu(adj @ T1 + b1)   (per-batch, grid.z = 32)
    gemm_bf16_wmma<float, unsigned short, true, true, true>
        <<<dim3(HDIM / 64, NNODES / 128, NBATCH), blk, 0, stream>>>(
        adj, NNODES, 0, P, HDIM, sAct, Q, HDIM, sAct, b1, NNODES);
    bn_kernel<<<dim3(HDIM, NBATCH), blk, 0, stream>>>(Q, HDIM);

    // ---- Layer 2 ----
    gemm_bf16_wmma<unsigned short, float, true, false, false>
        <<<dim3(HDIM / 64, M_all / 128, 1), blk, 0, stream>>>(
        Q, HDIM, 0, W2, HDIM, 0, P, HDIM, 0, nullptr, HDIM);
    gemm_bf16_wmma<float, unsigned short, true, true, true>
        <<<dim3(HDIM / 64, NNODES / 128, NBATCH), blk, 0, stream>>>(
        adj, NNODES, 0, P, HDIM, sAct, Q, HDIM, sAct, b2, NNODES);
    bn_kernel<<<dim3(HDIM, NBATCH), blk, 0, stream>>>(Q, HDIM);

    // ---- Layer 3 ----
    gemm_bf16_wmma<unsigned short, float, true, false, false>
        <<<dim3(CDIM / 64, M_all / 128, 1), blk, 0, stream>>>(
        Q, HDIM, 0, W3, CDIM, 0, P, CDIM, 0, nullptr, HDIM);
    gemm_bf16_wmma<float, unsigned short, true, true, true>
        <<<dim3(CDIM / 64, NNODES / 128, NBATCH), blk, 0, stream>>>(
        adj, NNODES, 0, P, CDIM, sC, Q, CDIM, sC, b3, NNODES);

    // ---- Masked pooling + MLP head ----
    count_kernel<<<1, blk, 0, stream>>>(x, denom);
    pool_kernel<<<dim3(128, NBATCH), blk, 0, stream>>>(Q, x, denom, pooled);
    head_kernel<<<NBATCH, blk, 0, stream>>>(pooled, LW1, Lb1, LW2, Lb2, LW3, Lb3, out);

    (void)in_sizes; (void)n_in; (void)out_size; (void)ws_size;
}
